// ExampleModel_71975061946546
// MI455X (gfx1250) — matrix operand order repaired and checked by reading, not compiled, not run on hardware
//
#include <hip/hip_runtime.h>
#include <hip/hip_bf16.h>
#include <math.h>

// ---------------- shapes ----------------
#define MOE_B 8
#define MOE_T 512
#define MOE_M 2048
#define MOE_H 1024
#define MOE_E 2
#define MOE_S (MOE_B * MOE_T)   // 4096

typedef __attribute__((ext_vector_type(16))) __bf16 v16bf;
typedef __attribute__((ext_vector_type(8)))  __bf16 v8bf;
typedef __attribute__((ext_vector_type(8)))  float  v8f;

union Frag16 { v16bf v; v8bf h[2]; };

// ---------------------------------------------------------------------------
// 1) split x (fp32) -> bf16 hi + bf16 lo residual
// ---------------------------------------------------------------------------
__global__ void moe_split_x(const float* __restrict__ x,
                            unsigned short* __restrict__ xh_u,
                            unsigned short* __restrict__ xl_u) {
  size_t i = (size_t)blockIdx.x * blockDim.x + threadIdx.x;
  __bf16* xh = (__bf16*)xh_u;
  __bf16* xl = (__bf16*)xl_u;
  float v = x[i];
  __bf16 hi = (__bf16)v;
  float r = v - (float)hi;
  xh[i] = hi;
  xl[i] = (__bf16)r;
}

// ---------------------------------------------------------------------------
// 2) transpose w1 [E][M][H] -> w1t [E][H][M], split into bf16 hi/lo
// ---------------------------------------------------------------------------
__global__ void moe_split_w1(const float* __restrict__ w1,
                             unsigned short* __restrict__ wh_u,
                             unsigned short* __restrict__ wl_u) {
  size_t i = (size_t)blockIdx.x * blockDim.x + threadIdx.x;
  __bf16* wh = (__bf16*)wh_u;
  __bf16* wl = (__bf16*)wl_u;
  int e = (int)(i / ((size_t)MOE_M * MOE_H));
  size_t rem = i - (size_t)e * MOE_M * MOE_H;
  int m = (int)(rem / MOE_H);
  int h = (int)(rem % MOE_H);
  float v = w1[i];
  __bf16 hi = (__bf16)v;
  float r = v - (float)hi;
  size_t o = ((size_t)e * MOE_H + h) * MOE_M + m;
  wh[o] = hi;
  wl[o] = (__bf16)r;
}

// ---------------------------------------------------------------------------
// 3) gate: logits = x·wg (fp32), gates = softmax over 2 experts
// ---------------------------------------------------------------------------
__global__ void moe_gate(const float* __restrict__ x,
                         const float* __restrict__ wg,
                         float* __restrict__ gates) {
  int wave = threadIdx.x >> 5;
  int lane = threadIdx.x & 31;
  int s = blockIdx.x * 8 + wave;
  const float* xs = x + (size_t)s * MOE_M;
  float a0 = 0.f, a1 = 0.f;
  for (int m = lane; m < MOE_M; m += 32) {
    float xv = xs[m];
    a0 += xv * wg[m * 2 + 0];
    a1 += xv * wg[m * 2 + 1];
  }
  for (int d = 1; d < 32; d <<= 1) {
    a0 += __shfl_xor(a0, d);
    a1 += __shfl_xor(a1, d);
  }
  if (lane == 0) {
    float g0 = 1.f / (1.f + expf(a1 - a0));
    gates[2 * s + 0] = g0;
    gates[2 * s + 1] = 1.f - g0;
  }
}

// ---------------------------------------------------------------------------
// 4) w2sum[e*H+h] = sum_m w2[e][h][m];  b2sum[e] = sum_m b2[e][m]
//    grid: 257 blocks of 256 (blocks 0..255 -> 2048 rows; block 256 -> b2)
// ---------------------------------------------------------------------------
__global__ void moe_w2sum(const float* __restrict__ w2,
                          const float* __restrict__ b2,
                          float* __restrict__ w2sum,
                          float* __restrict__ b2sum) {
  int wave = threadIdx.x >> 5;
  int lane = threadIdx.x & 31;
  if (blockIdx.x < 256) {
    int row = blockIdx.x * 8 + wave;            // 0 .. 2047  ( = e*H + h )
    const float* p = w2 + (size_t)row * MOE_M;
    float a = 0.f;
    for (int m = lane; m < MOE_M; m += 32) a += p[m];
    for (int d = 1; d < 32; d <<= 1) a += __shfl_xor(a, d);
    if (lane == 0) w2sum[row] = a;
  } else {
    if (wave < MOE_E) {
      const float* p = b2 + (size_t)wave * MOE_M;
      float a = 0.f;
      for (int m = lane; m < MOE_M; m += 32) a += p[m];
      for (int d = 1; d < 32; d <<= 1) a += __shfl_xor(a, d);
      if (lane == 0) b2sum[wave] = a;
    }
  }
}

// ---------------------------------------------------------------------------
// 5) fused expert layer-1 GEMM (bf16x3 WMMA) + relu + dot(w2sum) reduction
//    grid = (S/16, E); block = 256 (8 waves). Wave w owns h-tiles w, w+8, ...
//    partial[e*S + s] = sum_h relu(x_s·W1_e[:,h] + b1) * w2sum_e[h]
// ---------------------------------------------------------------------------
__global__ __launch_bounds__(256)
void moe_ffn1(const unsigned short* __restrict__ xh_u,
              const unsigned short* __restrict__ xl_u,
              const unsigned short* __restrict__ wh_u,
              const unsigned short* __restrict__ wl_u,
              const float* __restrict__ b1,
              const float* __restrict__ w2sum,
              float* __restrict__ partial) {
  const __bf16* xh = (const __bf16*)xh_u;
  const __bf16* xl = (const __bf16*)xl_u;
  const __bf16* wh = (const __bf16*)wh_u;
  const __bf16* wl = (const __bf16*)wl_u;

  const int e     = blockIdx.y;
  const int stile = blockIdx.x;
  const int wave  = threadIdx.x >> 5;
  const int lane  = threadIdx.x & 31;

  __shared__ float ldsPart[16];
  if (threadIdx.x < 16) ldsPart[threadIdx.x] = 0.f;
  __syncthreads();

  // A-fragment layout (16-bit A 16x32, ISA 7.12.2):
  //   lanes 0-15: row = lane,  K in {0..7} U {16..23}
  //   lanes16-31: row = lane-16, K in {8..15} U {24..31}
  const int arow = lane & 15;
  const int akb  = (lane < 16) ? 0 : 8;
  const __bf16* axh = xh + ((size_t)(stile * 16 + arow)) * MOE_M + akb;
  const __bf16* axl = xl + ((size_t)(stile * 16 + arow)) * MOE_M + akb;

  // B-fragment layout (16-bit B 32x16): lane n<16 -> col n, K in {0..7} U {16..23};
  // lanes 16-31 -> col n-16, K in {8..15} U {24..31}. w1t is [E][H][M] so K is contiguous.
  const int bcol  = lane & 15;
  const int bkb   = (lane < 16) ? 0 : 8;
  const size_t wbase = (size_t)e * MOE_H * MOE_M;

  v8f acc[8];
  const v8f vzero = {0.f, 0.f, 0.f, 0.f, 0.f, 0.f, 0.f, 0.f};
#pragma unroll
  for (int i = 0; i < 8; ++i) acc[i] = vzero;

  for (int kt = 0; kt < MOE_M / 32; ++kt) {
    const int k0 = kt * 32;
    Frag16 Ah, Al;
    Ah.h[0] = *(const v8bf*)(axh + k0);
    Ah.h[1] = *(const v8bf*)(axh + k0 + 16);
    Al.h[0] = *(const v8bf*)(axl + k0);
    Al.h[1] = *(const v8bf*)(axl + k0 + 16);
    __builtin_prefetch(axh + k0 + 32, 0, 0);   // global_prefetch_b8

#pragma unroll
    for (int i = 0; i < 8; ++i) {
      const int ht = wave + i * 8;             // 64 h-tiles across 8 waves
      const __bf16* pbh = wh + wbase + (size_t)(ht * 16 + bcol) * MOE_M + bkb + k0;
      const __bf16* pbl = wl + wbase + (size_t)(ht * 16 + bcol) * MOE_M + bkb + k0;
      Frag16 Bh, Bl;
      Bh.h[0] = *(const v8bf*)(pbh);
      Bh.h[1] = *(const v8bf*)(pbh + 16);
      Bl.h[0] = *(const v8bf*)(pbl);
      Bl.h[1] = *(const v8bf*)(pbl + 16);
      // bf16x3: xh*wh + xh*wl + xl*wh  (f32 accumulate)
      acc[i] = __builtin_amdgcn_wmma_f32_16x16x32_bf16(
          false, Ah.v, false, Bh.v, (short)0, acc[i], false, false);
      acc[i] = __builtin_amdgcn_wmma_f32_16x16x32_bf16(
          false, Ah.v, false, Bl.v, (short)0, acc[i], false, false);
      acc[i] = __builtin_amdgcn_wmma_f32_16x16x32_bf16(
          false, Al.v, false, Bh.v, (short)0, acc[i], false, false);
    }
  }

  // Epilogue: C layout (f32 16x16): VGPR r, lane L -> M = r + (L>=16 ? 8:0),
  // N = L&15.  val = relu(c + b1[h]) * w2sum[h]; reduce over 16 columns.
#pragma unroll
  for (int i = 0; i < 8; ++i) {
    const int ht = wave + i * 8;
    const int h = ht * 16 + (lane & 15);
    const float w2s = w2sum[e * MOE_H + h];
    const float bb = b1[e * MOE_H + h];
#pragma unroll
    for (int r = 0; r < 8; ++r) {
      float v = acc[i][r] + bb;
      v = fmaxf(v, 0.f) * w2s;
      v += __shfl_xor(v, 1);
      v += __shfl_xor(v, 2);
      v += __shfl_xor(v, 4);
      v += __shfl_xor(v, 8);
      if ((lane & 15) == 0) {
        const int srow = r + ((lane < 16) ? 0 : 8);
        atomicAdd(&ldsPart[srow], v);
      }
    }
  }
  __syncthreads();
  if (threadIdx.x < 16)
    partial[(size_t)e * MOE_S + stile * 16 + threadIdx.x] = ldsPart[threadIdx.x];
}

// ---------------------------------------------------------------------------
// 6) combine + per-batch log_softmax over T=512. grid = B, block = T.
// ---------------------------------------------------------------------------
__global__ void moe_combine_lsm(const float* __restrict__ gates,
                                const float* __restrict__ partial,
                                const float* __restrict__ b2sum,
                                float* __restrict__ out) {
  const int b = blockIdx.x;
  const int t = threadIdx.x;
  const int s = b * MOE_T + t;
  float val = gates[2 * s + 0] * (partial[s] + b2sum[0]) +
              gates[2 * s + 1] * (partial[MOE_S + s] + b2sum[1]);
  __shared__ float buf[MOE_T];
  buf[t] = val;
  __syncthreads();
  for (int off = MOE_T / 2; off >= 1; off >>= 1) {
    if (t < off) buf[t] = fmaxf(buf[t], buf[t + off]);
    __syncthreads();
  }
  const float mx = buf[0];
  __syncthreads();
  buf[t] = expf(val - mx);
  __syncthreads();
  for (int off = MOE_T / 2; off >= 1; off >>= 1) {
    if (t < off) buf[t] += buf[t + off];
    __syncthreads();
  }
  const float lse = mx + logf(buf[0]);
  out[s] = val - lse;
}

// ---------------------------------------------------------------------------
extern "C" void kernel_launch(void* const* d_in, const int* in_sizes, int n_in,
                              void* d_out, int out_size, void* d_ws, size_t ws_size,
                              hipStream_t stream) {
  const float* x  = (const float*)d_in[0];   // [B,T,M]
  const float* wg = (const float*)d_in[1];   // [M,E]
  const float* w1 = (const float*)d_in[2];   // [E,M,H]
  const float* b1 = (const float*)d_in[3];   // [E,H]
  const float* w2 = (const float*)d_in[4];   // [E,H,M]
  const float* b2 = (const float*)d_in[5];   // [E,M]
  float* out = (float*)d_out;                // [B,T]

  char* ws = (char*)d_ws;
  size_t off = 0;
  auto carve = [&](size_t bytes) {
    size_t o = off;
    off = (off + bytes + 255) & ~(size_t)255;
    return o;
  };
  unsigned short* xh   = (unsigned short*)(ws + carve((size_t)MOE_S * MOE_M * 2));
  unsigned short* xl   = (unsigned short*)(ws + carve((size_t)MOE_S * MOE_M * 2));
  unsigned short* w1h  = (unsigned short*)(ws + carve((size_t)MOE_E * MOE_M * MOE_H * 2));
  unsigned short* w1l  = (unsigned short*)(ws + carve((size_t)MOE_E * MOE_M * MOE_H * 2));
  float* w2sum   = (float*)(ws + carve((size_t)MOE_E * MOE_H * 4));
  float* b2sum   = (float*)(ws + carve((size_t)MOE_E * 4));
  float* gates   = (float*)(ws + carve((size_t)MOE_S * 2 * 4));
  float* partial = (float*)(ws + carve((size_t)MOE_E * MOE_S * 4));

  moe_split_x<<<(MOE_S * MOE_M) / 256, 256, 0, stream>>>(x, xh, xl);
  moe_split_w1<<<(MOE_E * MOE_M * MOE_H) / 256, 256, 0, stream>>>(w1, w1h, w1l);
  moe_gate<<<MOE_S / 8, 256, 0, stream>>>(x, wg, gates);
  moe_w2sum<<<257, 256, 0, stream>>>(w2, b2, w2sum, b2sum);
  moe_ffn1<<<dim3(MOE_S / 16, MOE_E), 256, 0, stream>>>(xh, xl, w1h, w1l, b1, w2sum, partial);
  moe_combine_lsm<<<MOE_B, MOE_T, 0, stream>>>(gates, partial, b2sum, out);
}
